// Block_52218212385395
// MI455X (gfx1250) — matrix operand order repaired and checked by reading, compiled, not run on hardware
//
#include <hip/hip_runtime.h>

// ---------------------------------------------------------------------------
// Types for CDNA5 WMMA (wave32): v_wmma_f32_16x16x32_bf16
// ---------------------------------------------------------------------------
typedef __attribute__((ext_vector_type(16))) __bf16        v16bf;
typedef __attribute__((ext_vector_type(8)))  float         v8f;
typedef __attribute__((ext_vector_type(8)))  unsigned int  v8u;
typedef __attribute__((ext_vector_type(4)))  unsigned int  v4u;

// Problem constants
#define BB   64
#define NN   196
#define CC   768
#define HEADS 12
#define HD   64
#define GG   8
#define CG   96
#define MLPD 3072
#define MROWS (BB * NN)     // 12544 = 98*128, divisible by 128
#define NPAD 224            // 196 padded up to 32-multiple for attention chunks

// ---------------------------------------------------------------------------
// A-fragment loader: 16-bit A matrix 16x32 (MxK) per ISA layout.
// lane l: row m = l%16; with kofs = (l>=16 ? 8 : 0) the lane's 16 elements are
// exactly the two contiguous 8-element runs  [kofs, kofs+8) and
// [kofs+16, kofs+24)  ->  two 16-byte loads (global_load_b128 / ds_load_b128).
// base_row points at row m of the operand; kb is the K offset (32-multiple).
// ---------------------------------------------------------------------------
__device__ __forceinline__ v16bf load_afrag(const __bf16* base_row, int lane, int kb) {
  const __bf16* p = base_row + kb + ((lane >= 16) ? 8 : 0);
  v4u lo = *(const v4u*)(p);
  v4u hi = *(const v4u*)(p + 16);
  v8u u;
  u[0] = lo[0]; u[1] = lo[1]; u[2] = lo[2]; u[3] = lo[3];
  u[4] = hi[0]; u[5] = hi[1]; u[6] = hi[2]; u[7] = hi[3];
  return __builtin_bit_cast(v16bf, u);
}

__device__ __forceinline__ float gelu_exact(float v) {
  return 0.5f * v * (1.0f + erff(v * 0.70710678118654752f));
}

// ---------------------------------------------------------------------------
// Block-tiled WMMA GEMM: out[m,n] = sum_k A[m,k] * Bt[k,n] (+bias,+gelu,+res)
//   Block: 256 threads (8 waves) -> 128x128 output tile.
//   Wave:  32x64 tile = 2 A-frags x 4 B-frags -> 8 WMMAs per 32-wide K step.
//   A tile (128x32 bf16, 8 KB) staged in LDS via GLOBAL_LOAD_ASYNC_TO_LDS_B128
//   (per-lane DMA, no dest VGPRs, tracked by ASYNCcnt); fragments rebuilt with
//   ds_load_b128 pairs.  Bt is pre-transposed [K][N] bf16 so B-fragments are
//   16 scalar loads per lane, strided by N (one column n).
// KV layout: write bf16 out as (b, col, n') with n'-stride 1, row pitch NPAD.
// ---------------------------------------------------------------------------
template <int STORE_BF16, int STORE_F32, int DO_GELU, int DO_RES, int KV>
__global__ __launch_bounds__(256) void gemm_kernel(
    const __bf16* __restrict__ A, const __bf16* __restrict__ Bt,
    const float* __restrict__ bias, const float* __restrict__ res,
    float* __restrict__ outf, __bf16* __restrict__ outh,
    int M, int N, int K) {
  __shared__ __bf16 Atile[128][32];

  int wave = threadIdx.x >> 5, lane = threadIdx.x & 31;
  int nblks = N >> 7;                 // N / 128
  int mtile = blockIdx.x / nblks;     // 128-row block
  int ntile = blockIdx.x % nblks;     // 128-col block
  int mw = wave & 3;                  // wave's 32-row sub-tile
  int nw = wave >> 2;                 // wave's 64-col sub-tile
  int half8 = (lane >= 16) ? 8 : 0;

  const __bf16* bbase =
      Bt + (size_t)half8 * N + ntile * 128 + nw * 64 + (lane & 15);  // col lane%16, k rows strided

  // Per-thread chunk assignment for the A tile (512 x 16B chunks, 2/thread).
  // Chunk c: row r = c>>2, k-offset ko = (c&3)*8.  LDS byte offsets are the
  // low 32 bits of the generic pointer (LDS aperture rule: addr[31:0]).
  int c0r = threadIdx.x >> 2, c0k = (threadIdx.x & 3) * 8;
  int c1r = (threadIdx.x + 256) >> 2, c1k = ((threadIdx.x + 256) & 3) * 8;
  unsigned lds0 = (unsigned)(unsigned long long)&Atile[c0r][c0k];
  unsigned lds1 = (unsigned)(unsigned long long)&Atile[c1r][c1k];
  const __bf16* gsrc0 = A + (size_t)(mtile * 128 + c0r) * K + c0k;
  const __bf16* gsrc1 = A + (size_t)(mtile * 128 + c1r) * K + c1k;

  v8f acc[2][4];
#pragma unroll
  for (int sm = 0; sm < 2; ++sm)
#pragma unroll
    for (int f = 0; f < 4; ++f) acc[sm][f] = v8f{};

  for (int kb = 0; kb < K; kb += 32) {
    // ---- async-DMA the A tile (128 rows x 32 k) into LDS ----
    __syncthreads();  // previous tile fully consumed by all waves
    asm volatile("global_load_async_to_lds_b128 %0, %1, off"
                 :: "v"(lds0), "v"(gsrc0 + kb) : "memory");
    asm volatile("global_load_async_to_lds_b128 %0, %1, off"
                 :: "v"(lds1), "v"(gsrc1 + kb) : "memory");
    asm volatile("s_wait_asynccnt 0" ::: "memory");  // this wave's DMA landed
    __syncthreads();                                 // all waves' DMA landed

    // ---- fragments + 8 WMMAs ----
    v16bf a0 = load_afrag(&Atile[mw * 32 + (lane & 15)][0], lane, 0);
    v16bf a1 = load_afrag(&Atile[mw * 32 + 16 + (lane & 15)][0], lane, 0);
    if (kb + 32 < K) __builtin_prefetch(bbase + (size_t)(kb + 32) * N, 0, 0);
#pragma unroll
    for (int f = 0; f < 4; ++f) {
      v16bf b;
#pragma unroll
      for (int s = 0; s < 16; ++s)
        b[s] = bbase[(size_t)(kb + s + (s >= 8 ? 8 : 0)) * N + f * 16];
      acc[0][f] = __builtin_amdgcn_wmma_f32_16x16x32_bf16(
          false, a0, false, b, (short)0, acc[0][f], false, false);
      acc[1][f] = __builtin_amdgcn_wmma_f32_16x16x32_bf16(
          false, a1, false, b, (short)0, acc[1][f], false, false);
    }
  }

  // ---- epilogue ----
#pragma unroll
  for (int sm = 0; sm < 2; ++sm) {
#pragma unroll
    for (int f = 0; f < 4; ++f) {
      int coln = ntile * 128 + nw * 64 + f * 16 + (lane & 15);
      float bv = bias ? bias[coln] : 0.0f;
#pragma unroll
      for (int r = 0; r < 8; ++r) {
        int row = mtile * 128 + mw * 32 + sm * 16 + r + half8;
        float v = acc[sm][f][r] + bv;
        if (DO_GELU) v = gelu_exact(v);
        if (DO_RES) v += res[(size_t)row * N + coln];
        if (STORE_F32) outf[(size_t)row * N + coln] = v;
        if (STORE_BF16) {
          if (KV) {
            int b_ = row / NN, np = row % NN;
            outh[((size_t)b_ * CC + coln) * NPAD + np] = (__bf16)v;
          } else {
            outh[(size_t)row * N + coln] = (__bf16)v;
          }
        }
      }
    }
  }
}

// ---------------------------------------------------------------------------
// LayerNorm over C=768, one wave32 per row, write bf16.
// ---------------------------------------------------------------------------
__global__ __launch_bounds__(256) void ln_kernel(
    const float* __restrict__ x, const float* __restrict__ g,
    const float* __restrict__ b, __bf16* __restrict__ out, int rows, int C,
    float eps) {
  int wave = threadIdx.x >> 5, lane = threadIdx.x & 31;
  int row = blockIdx.x * 8 + wave;
  if (row >= rows) return;
  const float* xr = x + (size_t)row * C;
  float s = 0.f, s2 = 0.f;
  for (int c = lane; c < C; c += 32) {
    float v = xr[c];
    s += v;
    s2 += v * v;
  }
#pragma unroll
  for (int m = 1; m <= 16; m <<= 1) {
    s += __shfl_xor(s, m, 32);
    s2 += __shfl_xor(s2, m, 32);
  }
  float mu = s / C;
  float var = s2 / C - mu * mu;
  float rstd = rsqrtf(var + eps);
  __bf16* orow = out + (size_t)row * C;
  for (int c = lane; c < C; c += 32)
    orow[c] = (__bf16)((xr[c] - mu) * rstd * g[c] + b[c]);
}

// ---------------------------------------------------------------------------
// Weight transpose fp32 [R][Cc] -> bf16 [Cc][R]
// ---------------------------------------------------------------------------
__global__ void transpose_bf16(const float* __restrict__ w,
                               __bf16* __restrict__ wt, int R, int Cc) {
  size_t i = (size_t)blockIdx.x * blockDim.x + threadIdx.x;
  if (i >= (size_t)R * Cc) return;
  int r = (int)(i / Cc), c = (int)(i % Cc);
  wt[(size_t)c * R + r] = (__bf16)w[i];
}

// ---------------------------------------------------------------------------
// Depthwise 3x3 conv on q (B,C,H,W view of row-major (B,N,C) bf16) + bias.
// Output layout [bg][n][cg] fp32.
// ---------------------------------------------------------------------------
__global__ void dwconv_kernel(const __bf16* __restrict__ q,
                              const float* __restrict__ w,
                              const float* __restrict__ bias,
                              float* __restrict__ o) {
  int i = blockIdx.x * blockDim.x + threadIdx.x;
  if (i >= BB * GG * NN * CG) return;
  int cg = i % CG;
  int rem = i / CG;
  int n = rem % NN;
  int bg = rem / NN;
  int y = n / 14, x = n % 14;
  int b = bg / GG, g = bg % GG;
  int ch = g * CG + cg;
  float acc = bias[cg];
#pragma unroll
  for (int ky = 0; ky < 3; ++ky) {
    int yy = y + ky - 1;
    if (yy < 0 || yy > 13) continue;
#pragma unroll
    for (int kx = 0; kx < 3; ++kx) {
      int xx = x + kx - 1;
      if (xx < 0 || xx > 13) continue;
      float qv = (float)q[((size_t)(b * NN + yy * 14 + xx)) * CC + ch];
      acc += qv * w[cg * 9 + ky * 3 + kx];
    }
  }
  o[i] = acc;
}

// ---------------------------------------------------------------------------
// Offset head: LN(96) -> exact GELU -> proj(2) -> tanh -> + reference grid.
// Emits pixel-space sample coords (gx, gy) per (bg, n).
// ---------------------------------------------------------------------------
__global__ void offset_kernel(const float* __restrict__ o,
                              const float* __restrict__ g,
                              const float* __restrict__ b,
                              const float* __restrict__ proj,
                              float* __restrict__ pos) {
  int i = blockIdx.x * blockDim.x + threadIdx.x;
  if (i >= BB * GG * NN) return;
  const float* row = o + (size_t)i * CG;
  float s = 0.f, s2 = 0.f;
  for (int c = 0; c < CG; ++c) {
    float v = row[c];
    s += v;
    s2 += v * v;
  }
  float mu = s / CG;
  float var = s2 / CG - mu * mu;
  float rstd = rsqrtf(var + 1e-5f);
  float a0 = 0.f, a1 = 0.f;
  for (int c = 0; c < CG; ++c) {
    float t = (row[c] - mu) * rstd * g[c] + b[c];
    float ge = gelu_exact(t);
    a0 += proj[c] * ge;        // offset[...,0] (y)
    a1 += proj[CG + c] * ge;   // offset[...,1] (x)
  }
  int n = i % NN;
  int y = n / 14, x = n % 14;
  float offy = tanhf(a0) * (1.0f / 14.0f);
  float offx = tanhf(a1) * (1.0f / 14.0f);
  float refy = (y + 0.5f) / 14.0f * 2.0f - 1.0f;
  float refx = (x + 0.5f) / 14.0f * 2.0f - 1.0f;
  float py = offy + refy, px = offx + refx;  // grid = pos[..., ::-1]
  pos[(size_t)i * 2 + 0] = (px + 1.0f) * 0.5f * 13.0f;  // gx (pixels)
  pos[(size_t)i * 2 + 1] = (py + 1.0f) * 0.5f * 13.0f;  // gy (pixels)
}

// ---------------------------------------------------------------------------
// Bilinear grid sample of xn (B,N,C bf16) -> sampled (B,N,C bf16).
// ---------------------------------------------------------------------------
__global__ void gridsample_kernel(const __bf16* __restrict__ xn,
                                  const float* __restrict__ pos,
                                  __bf16* __restrict__ out) {
  size_t i = (size_t)blockIdx.x * blockDim.x + threadIdx.x;
  if (i >= (size_t)MROWS * CC) return;
  int c = (int)(i % CC);
  size_t row = i / CC;
  int b = (int)(row / NN), n = (int)(row % NN);
  int g = c / CG;
  int bg = b * GG + g;
  float gx = pos[((size_t)bg * NN + n) * 2 + 0];
  float gy = pos[((size_t)bg * NN + n) * 2 + 1];
  float x0f = floorf(gx), y0f = floorf(gy);
  int x0 = (int)x0f, y0 = (int)y0f;
  float wx1 = gx - x0f, wx0 = 1.0f - wx1;
  float wy1 = gy - y0f, wy0 = 1.0f - wy1;
  auto fetch = [&](int xi, int yi) -> float {
    if (xi < 0 || xi > 13 || yi < 0 || yi > 13) return 0.0f;
    return (float)xn[((size_t)(b * NN) + yi * 14 + xi) * CC + c];
  };
  float v = fetch(x0, y0) * wx0 * wy0 + fetch(x0, y0 + 1) * wx0 * wy1 +
            fetch(x0 + 1, y0) * wx1 * wy0 + fetch(x0 + 1, y0 + 1) * wx1 * wy1;
  out[i] = (__bf16)v;
}

// ---------------------------------------------------------------------------
// Flash-style attention. One wave per (b, head, 16-row Q tile).
//   q  : (B,N,C) bf16 row-major         (A operand of QK^T)
//   kcn: (B,C,NPAD) bf16, K^T layout    (B operand of QK^T: [d][n'])
//   v  : (B,N,C) bf16 row-major         (B operand of AV:   [n'][d])
//   out: (B,N,C) bf16 row-major
// Online softmax in fp32 (scale = HD^-0.5 = 0.125), P staged via LDS to
// convert D-fragment layout into A-fragment layout.
// ---------------------------------------------------------------------------
__global__ __launch_bounds__(256) void attn_kernel(
    const __bf16* __restrict__ q, const __bf16* __restrict__ kcn,
    const __bf16* __restrict__ v, __bf16* __restrict__ out) {
  __shared__ __bf16 plds[8][16][32];
  int w = threadIdx.x >> 5, lane = threadIdx.x & 31;
  int gw = blockIdx.x * 8 + w;            // 0 .. B*HEADS*13-1 (exact grid)
  int mtile = gw % 13;
  int bh = gw / 13;
  int b = bh / HEADS, h = bh % HEADS;
  int half8 = (lane >= 16) ? 8 : 0;

  // Q fragments (16 rows x 64 d), rows clamped into [0,195]
  int mq = mtile * 16 + (lane & 15);
  if (mq > NN - 1) mq = NN - 1;
  const __bf16* qrow = q + ((size_t)(b * NN + mq)) * CC + h * HD;
  v16bf aq0 = load_afrag(qrow, lane, 0);
  v16bf aq1 = load_afrag(qrow, lane, 32);

  float mi[8], li[8];
#pragma unroll
  for (int r = 0; r < 8; ++r) { mi[r] = -1e30f; li[r] = 0.0f; }
  v8f oacc[4] = {v8f{}, v8f{}, v8f{}, v8f{}};

  const __bf16* kbase = kcn + ((size_t)b * CC + h * HD) * NPAD;

  for (int chunk = 0; chunk < 7; ++chunk) {
    int n0 = chunk * 32;
    // QK^T for 32 key columns: two 16x16 score frags, k-dim = 64 (2 steps)
    v8f s0 = {}, s1 = {};
    {
      v16bf bk;
      for (int s = 0; s < 16; ++s) bk[s] = kbase[(size_t)(half8 + s + (s >= 8 ? 8 : 0)) * NPAD + n0 + (lane & 15)];
      s0 = __builtin_amdgcn_wmma_f32_16x16x32_bf16(false, aq0, false, bk,
                                                   (short)0, s0, false, false);
      for (int s = 0; s < 16; ++s) bk[s] = kbase[(size_t)(32 + half8 + s + (s >= 8 ? 8 : 0)) * NPAD + n0 + (lane & 15)];
      s0 = __builtin_amdgcn_wmma_f32_16x16x32_bf16(false, aq1, false, bk,
                                                   (short)0, s0, false, false);
      for (int s = 0; s < 16; ++s) bk[s] = kbase[(size_t)(half8 + s + (s >= 8 ? 8 : 0)) * NPAD + n0 + 16 + (lane & 15)];
      s1 = __builtin_amdgcn_wmma_f32_16x16x32_bf16(false, aq0, false, bk,
                                                   (short)0, s1, false, false);
      for (int s = 0; s < 16; ++s) bk[s] = kbase[(size_t)(32 + half8 + s + (s >= 8 ? 8 : 0)) * NPAD + n0 + 16 + (lane & 15)];
      s1 = __builtin_amdgcn_wmma_f32_16x16x32_bf16(false, aq1, false, bk,
                                                   (short)0, s1, false, false);
    }
    // Online softmax per row (rows live in reg r, replicated across 16-lane half)
    int c0 = n0 + (lane & 15);
#pragma unroll
    for (int r = 0; r < 8; ++r) {
      float f0 = s0[r] * 0.125f;
      float f1 = s1[r] * 0.125f;
      if (c0 >= NN) f0 = -1e30f;
      if (c0 + 16 >= NN) f1 = -1e30f;
      float t = fmaxf(f0, f1);
      t = fmaxf(t, __shfl_xor(t, 1, 32));
      t = fmaxf(t, __shfl_xor(t, 2, 32));
      t = fmaxf(t, __shfl_xor(t, 4, 32));
      t = fmaxf(t, __shfl_xor(t, 8, 32));
      float mnew = fmaxf(mi[r], t);
      float alpha = __expf(mi[r] - mnew);
      float p0 = __expf(f0 - mnew);
      float p1 = __expf(f1 - mnew);
      float rs = p0 + p1;
      rs += __shfl_xor(rs, 1, 32);
      rs += __shfl_xor(rs, 2, 32);
      rs += __shfl_xor(rs, 4, 32);
      rs += __shfl_xor(rs, 8, 32);
      li[r] = li[r] * alpha + rs;
      mi[r] = mnew;
      oacc[0][r] *= alpha;
      oacc[1][r] *= alpha;
      oacc[2][r] *= alpha;
      oacc[3][r] *= alpha;
      int mrow = r + half8;
      plds[w][mrow][lane & 15] = (__bf16)p0;
      plds[w][mrow][16 + (lane & 15)] = (__bf16)p1;
    }
    asm volatile("s_wait_dscnt 0" ::: "memory");
    // Rebuild P as an A-fragment (16x32) from LDS (two ds_load_b128 per lane)
    v16bf pa = load_afrag(&plds[w][lane & 15][0], lane, 0);
    // AV: V chunk is 32 keys x 64 d; masked keys have P ~ 0
    const __bf16* vrow = v + ((size_t)(b * NN)) * CC + h * HD + (lane & 15);
#pragma unroll
    for (int f = 0; f < 4; ++f) {
      v16bf bv;
      for (int s = 0; s < 16; ++s) {
        int np = n0 + half8 + s + (s >= 8 ? 8 : 0);
        if (np > NN - 1) np = NN - 1;
        bv[s] = vrow[(size_t)np * CC + f * 16];
      }
      oacc[f] = __builtin_amdgcn_wmma_f32_16x16x32_bf16(
          false, pa, false, bv, (short)0, oacc[f], false, false);
    }
  }

  // Normalize and store
#pragma unroll
  for (int r = 0; r < 8; ++r) {
    int m = mtile * 16 + r + half8;
    if (m >= NN) continue;
    float inv = 1.0f / li[r];
#pragma unroll
    for (int f = 0; f < 4; ++f) {
      int d = h * HD + f * 16 + (lane & 15);
      out[((size_t)(b * NN + m)) * CC + d] = (__bf16)(oacc[f][r] * inv);
    }
  }
}

// ---------------------------------------------------------------------------
// Host orchestration
// ---------------------------------------------------------------------------
extern "C" void kernel_launch(void* const* d_in, const int* in_sizes, int n_in,
                              void* d_out, int out_size, void* d_ws,
                              size_t ws_size, hipStream_t stream) {
  const float* x        = (const float*)d_in[0];
  const float* ln1_g    = (const float*)d_in[1];
  const float* ln1_b    = (const float*)d_in[2];
  const float* ln2_g    = (const float*)d_in[3];
  const float* ln2_b    = (const float*)d_in[4];
  const float* Wq       = (const float*)d_in[5];
  const float* bq       = (const float*)d_in[6];
  const float* Wk       = (const float*)d_in[7];
  const float* bk       = (const float*)d_in[8];
  const float* Wv       = (const float*)d_in[9];
  const float* bv       = (const float*)d_in[10];
  const float* Wo       = (const float*)d_in[11];
  const float* bo       = (const float*)d_in[12];
  const float* off_dw_w = (const float*)d_in[13];
  const float* off_dw_b = (const float*)d_in[14];
  const float* off_ln_g = (const float*)d_in[15];
  const float* off_ln_b = (const float*)d_in[16];
  const float* off_proj = (const float*)d_in[17];
  const float* W1       = (const float*)d_in[18];
  const float* b1       = (const float*)d_in[19];
  const float* W2       = (const float*)d_in[20];
  const float* b2       = (const float*)d_in[21];
  float* outp = (float*)d_out;

  char* wsp = (char*)d_ws;
  size_t off = 0;
  auto alloc = [&](size_t bytes) -> void* {
    void* p = wsp + off;
    off += (bytes + 255) & ~(size_t)255;
    return p;
  };

  __bf16* xn   = (__bf16*)alloc((size_t)MROWS * CC * 2);
  __bf16* WqT  = (__bf16*)alloc((size_t)CC * CC * 2);
  __bf16* WkT  = (__bf16*)alloc((size_t)CC * CC * 2);
  __bf16* WvT  = (__bf16*)alloc((size_t)CC * CC * 2);
  __bf16* WoT  = (__bf16*)alloc((size_t)CC * CC * 2);
  __bf16* W1T  = (__bf16*)alloc((size_t)CC * MLPD * 2);
  __bf16* W2T  = (__bf16*)alloc((size_t)MLPD * CC * 2);
  __bf16* qb   = (__bf16*)alloc((size_t)MROWS * CC * 2);
  float*  oact = (float*)alloc((size_t)BB * GG * NN * CG * 4);
  float*  pos  = (float*)alloc((size_t)BB * GG * NN * 2 * 4);
  __bf16* samp = (__bf16*)alloc((size_t)MROWS * CC * 2);
  __bf16* kcn  = (__bf16*)alloc((size_t)BB * CC * NPAD * 2);
  __bf16* vrm  = (__bf16*)alloc((size_t)MROWS * CC * 2);
  __bf16* atto = (__bf16*)alloc((size_t)MROWS * CC * 2);
  float*  x2   = (float*)alloc((size_t)MROWS * CC * 4);
  __bf16* xn2  = (__bf16*)alloc((size_t)MROWS * CC * 2);
  __bf16* mlp  = (__bf16*)alloc((size_t)MROWS * MLPD * 2);

  const int LNB = MROWS / 8;  // 1568 blocks, 8 waves each
  ln_kernel<<<LNB, 256, 0, stream>>>(x, ln1_g, ln1_b, xn, MROWS, CC, 1e-6f);

  transpose_bf16<<<(CC * CC + 255) / 256, 256, 0, stream>>>(Wq, WqT, CC, CC);
  transpose_bf16<<<(CC * CC + 255) / 256, 256, 0, stream>>>(Wk, WkT, CC, CC);
  transpose_bf16<<<(CC * CC + 255) / 256, 256, 0, stream>>>(Wv, WvT, CC, CC);
  transpose_bf16<<<(CC * CC + 255) / 256, 256, 0, stream>>>(Wo, WoT, CC, CC);
  transpose_bf16<<<(MLPD * CC + 255) / 256, 256, 0, stream>>>(W1, W1T, MLPD, CC);
  transpose_bf16<<<(CC * MLPD + 255) / 256, 256, 0, stream>>>(W2, W2T, CC, MLPD);

  const int GEMM_C_BLOCKS = (MROWS / 128) * (CC / 128);   // 98*6  = 588
  const int GEMM_M_BLOCKS = (MROWS / 128) * (MLPD / 128); // 98*24 = 2352

  // Q = LN1(x) @ Wq^T + bq  -> bf16 (B,N,C)
  gemm_kernel<1, 0, 0, 0, 0><<<GEMM_C_BLOCKS, 256, 0, stream>>>(
      xn, WqT, bq, nullptr, nullptr, qb, MROWS, CC, CC);

  // Offset network + deformed sampling
  dwconv_kernel<<<(BB * GG * NN * CG + 255) / 256, 256, 0, stream>>>(
      qb, off_dw_w, off_dw_b, oact);
  offset_kernel<<<(BB * GG * NN + 255) / 256, 256, 0, stream>>>(
      oact, off_ln_g, off_ln_b, off_proj, pos);
  gridsample_kernel<<<((size_t)MROWS * CC + 255) / 256, 256, 0, stream>>>(
      xn, pos, samp);

  // K (stored K^T layout) and V projections of sampled features
  gemm_kernel<1, 0, 0, 0, 1><<<GEMM_C_BLOCKS, 256, 0, stream>>>(
      samp, WkT, bk, nullptr, nullptr, kcn, MROWS, CC, CC);
  gemm_kernel<1, 0, 0, 0, 0><<<GEMM_C_BLOCKS, 256, 0, stream>>>(
      samp, WvT, bv, nullptr, nullptr, vrm, MROWS, CC, CC);

  // Attention: B*HEADS*13 row-tiles = 9984 waves = 1248 blocks of 8 waves
  attn_kernel<<<BB * HEADS * 13 / 8, 256, 0, stream>>>(qb, kcn, vrm, atto);

  // Output projection + residual -> x2 (fp32)
  gemm_kernel<0, 1, 0, 1, 0><<<GEMM_C_BLOCKS, 256, 0, stream>>>(
      atto, WoT, bo, x, x2, nullptr, MROWS, CC, CC);

  // LN2 -> MLP (GELU) -> residual
  ln_kernel<<<LNB, 256, 0, stream>>>(x2, ln2_g, ln2_b, xn2, MROWS, CC, 1e-6f);
  gemm_kernel<1, 0, 1, 0, 0><<<GEMM_M_BLOCKS, 256, 0, stream>>>(
      xn2, W1T, b1, nullptr, nullptr, mlp, MROWS, MLPD, CC);
  gemm_kernel<0, 1, 0, 1, 0><<<GEMM_C_BLOCKS, 256, 0, stream>>>(
      mlp, W2T, b2, x2, outp, nullptr, MROWS, CC, MLPD);
}
